// SwinTransformerBlock_8624294330870
// MI455X (gfx1250) — matrix operand order repaired and checked by reading, not compiled, not run on hardware
//
#include <hip/hip_runtime.h>
#include <hip/hip_bf16.h>
#include <math.h>

typedef __attribute__((ext_vector_type(16))) __bf16 v16bf;
typedef __attribute__((ext_vector_type(8)))  __bf16 v8bf;
typedef __attribute__((ext_vector_type(8)))  float  v8f;

#define BB     16
#define HH_    64
#define WW_    64
#define WS_    8
#define SHIFT_ 4
#define DIM_   256
#define NH_    8
#define HD_    32
#define HID_   1024
#define N_     64
#define NW_    64
#define NWIN_  (BB*NW_)     /* 1024 windows  */
#define TOK_   (BB*HH_*WW_) /* 65536 tokens  */

// ---------------------------------------------------------------------------
// Fragment loaders (per CDNA5 ISA §7.12.2 VGPR layouts, wave32)
// A (16x32 bf16): lane L holds row M=L&15; chunks k0=8*(L>>4) and k0+16.
// B (32x16 bf16): lane L holds col N=L&15; K chunks k0=8*(L>>4) and k0+16.
// C/D (16x16 f32): lane L, VGPR r -> row 8*(L>>4)+r, col L&15.
// ---------------------------------------------------------------------------
static __device__ inline v16bf ld_afrag(const __bf16* rowptr, int k0) {
  v8bf lo = *(const v8bf*)(rowptr + k0);
  v8bf hi = *(const v8bf*)(rowptr + k0 + 16);
  v16bf r;
#pragma unroll
  for (int i = 0; i < 8; ++i) { r[i] = lo[i]; r[i + 8] = hi[i]; }
  return r;
}

static __device__ inline v16bf ld_bfrag(const __bf16* p, int k0, int ld) {
  v16bf r;
#pragma unroll
  for (int i = 0; i < 8; ++i) { r[i] = p[(size_t)(k0 + i) * ld]; r[i + 8] = p[(size_t)(k0 + 16 + i) * ld]; }
  return r;
}

static __device__ inline v8f wmma_bf16(v16bf a, v16bf b, v8f c) {
  return __builtin_amdgcn_wmma_f32_16x16x32_bf16(
      /*neg_a=*/false, a, /*neg_b=*/false, b,
      /*c_mod=*/(short)0, c, /*reuse_a=*/false, /*reuse_b=*/false);
}

static __device__ inline int region9(int v) { return v < 56 ? 0 : (v < 60 ? 1 : 2); }

// ---------------------------------------------------------------------------
// K0: continuous position bias table -> bias16[NH][64*64] = 16*sigmoid(cpb)
// ---------------------------------------------------------------------------
static __device__ inline float cpb_coord(int a) {
  float v = (float)(a - 7) * (8.0f / 7.0f);
  float s = (v > 0.f) ? 1.f : ((v < 0.f) ? -1.f : 0.f);
  return s * log2f(fabsf(v) + 1.0f) * (1.0f / 3.0f); // /log2(8)
}

__global__ __launch_bounds__(256) void cpb_kernel(
    const float* __restrict__ w1, const float* __restrict__ b1,
    const float* __restrict__ w2, float* __restrict__ bias16) {
  __shared__ float table[225][NH_];
  int t = threadIdx.x;
  if (t < 225) {
    float t0 = cpb_coord(t / 15), t1 = cpb_coord(t % 15);
    float acc[NH_] = {};
    for (int k = 0; k < 512; ++k) {
      float h = fmaxf(t0 * w1[2 * k] + t1 * w1[2 * k + 1] + b1[k], 0.f);
#pragma unroll
      for (int n = 0; n < NH_; ++n) acc[n] += h * w2[n * 512 + k];
    }
#pragma unroll
    for (int n = 0; n < NH_; ++n) table[t][n] = acc[n];
  }
  __syncthreads();
  for (int o = t; o < NH_ * N_ * N_; o += 256) {
    int h = o >> 12, ij = o & 4095;
    int i = ij >> 6, j = ij & 63;
    int dy = (i >> 3) - (j >> 3) + 7;
    int dx = (i & 7) - (j & 7) + 7;
    float bv = table[dy * 15 + dx][h];
    bias16[o] = 16.0f / (1.0f + __expf(-bv));
  }
}

// ---------------------------------------------------------------------------
// K1: pack weights -> bf16 [K][N] (transposed), qkv bias vector, head scales
// ---------------------------------------------------------------------------
__global__ __launch_bounds__(256) void pack_kernel(
    const float* __restrict__ qkv_w, const float* __restrict__ proj_w,
    const float* __restrict__ mlp_w1, const float* __restrict__ mlp_w2,
    const float* __restrict__ q_bias, const float* __restrict__ v_bias,
    const float* __restrict__ logit_scale,
    __bf16* __restrict__ Bqkv, __bf16* __restrict__ Bproj,
    __bf16* __restrict__ Bm1, __bf16* __restrict__ Bm2,
    float* __restrict__ qkvb, float* __restrict__ scale) {
  int id = blockIdx.x * 256 + threadIdx.x;
  if (id < 196608) {                       // [256][768]
    int k = id / 768, n = id % 768;
    Bqkv[id] = (__bf16)qkv_w[n * 256 + k];
  } else if (id < 262144) {                // [256][256]
    int e = id - 196608; int k = e >> 8, n = e & 255;
    Bproj[e] = (__bf16)proj_w[n * 256 + k];
  } else if (id < 524288) {                // [256][1024]
    int e = id - 262144; int k = e >> 10, n = e & 1023;
    Bm1[e] = (__bf16)mlp_w1[n * 256 + k];
  } else if (id < 786432) {                // [1024][256]
    int e = id - 524288; int k = e >> 8, n = e & 255;
    Bm2[e] = (__bf16)mlp_w2[n * 1024 + k];
  }
  if (id < 768) qkvb[id] = (id < 256) ? q_bias[id] : (id < 512 ? 0.f : v_bias[id - 512]);
  if (id < NH_) scale[id] = fminf(__expf(logit_scale[id]), 100.0f);
}

// ---------------------------------------------------------------------------
// K2: cyclic shift (-4,-4) + window partition + bf16 convert
// ---------------------------------------------------------------------------
__global__ __launch_bounds__(256) void shiftwin_kernel(
    const float* __restrict__ x, __bf16* __restrict__ xw) {
  size_t idx = (size_t)blockIdx.x * 256 + threadIdx.x; // win*64*256 space
  int c = (int)(idx & 255);
  int row = (int)(idx >> 8);
  int tok = row & 63, win = row >> 6;
  int bi = win >> 6, w = win & 63;
  int y = ((w >> 3) * 8 + (tok >> 3) + SHIFT_) & 63;
  int xx = ((w & 7) * 8 + (tok & 7) + SHIFT_) & 63;
  xw[idx] = (__bf16)x[((size_t)bi * 4096 + y * 64 + xx) * DIM_ + c];
}

// ---------------------------------------------------------------------------
// Generic WMMA GEMM: C[M,N] = A[M,K](bf16,row) @ B[K,N](bf16,row) + bias
// block = 256 thr = 8 waves, block tile 64x128, wave tile 16x64, K step 32.
// EPI: 0 = f32 store, 2 = exact-GELU -> bf16 store, 3 = proj scatter
//      (window reverse + roll(+4,+4)) f32 store.
// ---------------------------------------------------------------------------
template <int EPI>
__global__ __launch_bounds__(256) void gemm_kernel(
    const __bf16* __restrict__ A, const __bf16* __restrict__ Bm,
    const float* __restrict__ bias, float* __restrict__ Cf,
    __bf16* __restrict__ Cb, int M, int N, int K, int nbn) {
  int lane = threadIdx.x & 31;
  int wave = threadIdx.x >> 5;
  int bm = blockIdx.x / nbn, bn = blockIdx.x % nbn;
  int m0 = bm * 64 + (wave >> 1) * 16;
  int n0 = bn * 128 + (wave & 1) * 64;
  int koff = (lane >> 4) * 8;
  const __bf16* arow = A + (size_t)(m0 + (lane & 15)) * K;

  v8f acc[4] = {};
  for (int k0 = 0; k0 < K; k0 += 32) {
    v16bf a = ld_afrag(arow + k0, koff);
    const __bf16* brow = Bm + (size_t)k0 * N + n0 + (lane & 15);
    if (k0 + 32 < K) __builtin_prefetch(arow + k0 + 32, 0, 0);
#pragma unroll
    for (int t = 0; t < 4; ++t) {
      v16bf b = ld_bfrag(brow + t * 16, koff, N);
      acc[t] = wmma_bf16(a, b, acc[t]);
    }
  }

  int hh = lane >> 4, cl = lane & 15;
#pragma unroll
  for (int t = 0; t < 4; ++t) {
    int col = n0 + t * 16 + cl;
    float bv = bias ? bias[col] : 0.f;
#pragma unroll
    for (int r = 0; r < 8; ++r) {
      int row = m0 + hh * 8 + r;
      float v = acc[t][r] + bv;
      if (EPI == 0) {
        Cf[(size_t)row * N + col] = v;
      } else if (EPI == 2) {
        float g = 0.5f * v * (1.0f + erff(v * 0.70710678118654752f));
        Cb[(size_t)row * N + col] = (__bf16)g;
      } else if (EPI == 3) {
        int win = row >> 6, tk = row & 63;
        int bi = win >> 6, w = win & 63;
        int y2 = (w >> 3) * 8 + (tk >> 3);
        int x2 = (w & 7) * 8 + (tk & 7);
        int y = (y2 + SHIFT_) & 63, xx = (x2 + SHIFT_) & 63;
        Cf[((size_t)bi * 4096 + y * 64 + xx) * DIM_ + col] = v;
      }
    }
  }
}

// ---------------------------------------------------------------------------
// K4: l2-normalize q/k per (head,token), pack qn[win][h][64][32],
//     knT[win][h][32][64] (transposed so QK^T B-frags are contiguous),
//     vp[win][h][64][32] -- all bf16.
// ---------------------------------------------------------------------------
__global__ __launch_bounds__(256) void normpack_kernel(
    const float* __restrict__ qkv, __bf16* __restrict__ qn,
    __bf16* __restrict__ knT, __bf16* __restrict__ vp) {
  int id = blockIdx.x * 256 + threadIdx.x; // NWIN_*NH_*N_ = 524288
  int tok = id & 63, h = (id >> 6) & 7, win = id >> 9;
  const float* rq = qkv + ((size_t)win * 64 + tok) * 768 + h * HD_;
  const float* rk = rq + 256;
  const float* rv = rq + 512;
  float qv[HD_], kv[HD_], sq = 0.f, sk = 0.f;
#pragma unroll
  for (int d = 0; d < HD_; ++d) {
    qv[d] = rq[d]; kv[d] = rk[d];
    sq += qv[d] * qv[d]; sk += kv[d] * kv[d];
  }
  float iq = 1.0f / fmaxf(sqrtf(sq), 1e-12f);
  float ik = 1.0f / fmaxf(sqrtf(sk), 1e-12f);
  size_t base = (size_t)win * NH_ + h;
  __bf16* qrow = qn + (base * 64 + tok) * HD_;
  __bf16* vrow = vp + (base * 64 + tok) * HD_;
  __bf16* kcol = knT + base * HD_ * 64 + tok;
#pragma unroll
  for (int d = 0; d < HD_; ++d) {
    qrow[d] = (__bf16)(qv[d] * iq);
    vrow[d] = (__bf16)rv[d];
    kcol[(size_t)d * 64] = (__bf16)(kv[d] * ik);
  }
}

// ---------------------------------------------------------------------------
// K5: fused window attention. 1 block / window, 1 wave / head.
//   QK^T: 16 WMMAs (K=32=HD in one shot); scale + CPB bias + shift mask;
//   row softmax via half-group shuffles; probs -> LDS (bf16);
//   P@V: 16 WMMAs; write attnout bf16 [win][64][256].
// ---------------------------------------------------------------------------
__global__ __launch_bounds__(256) void attn_kernel(
    const __bf16* __restrict__ qn, const __bf16* __restrict__ knT,
    const __bf16* __restrict__ vp, const float* __restrict__ bias16,
    const float* __restrict__ scale, __bf16* __restrict__ attnout) {
  extern __shared__ __bf16 probs[]; // NH_ * 64 * 64 bf16 = 64 KB
  int win = blockIdx.x;
  int h = threadIdx.x >> 5;
  int lane = threadIdx.x & 31;
  int hh = lane >> 4, cl = lane & 15;
  int koff = hh * 8;
  size_t base = (size_t)win * NH_ + h;

  // ---- QK^T ----
  v16bf aq[4];
#pragma unroll
  for (int ti = 0; ti < 4; ++ti)
    aq[ti] = ld_afrag(qn + (base * 64 + ti * 16 + cl) * HD_, koff);

  v8f acc[4][4] = {};
#pragma unroll
  for (int tj = 0; tj < 4; ++tj) {
    v16bf bk = ld_bfrag(knT + base * HD_ * 64 + tj * 16 + cl, koff, 64);
#pragma unroll
    for (int ti = 0; ti < 4; ++ti)
      acc[ti][tj] = wmma_bf16(aq[ti], bk, acc[ti][tj]);
  }

  // ---- scale + bias + mask + softmax ----
  float scl = scale[h];
  int winimg = win & 63;
  int whY = (winimg >> 3) * 8, wwX = (winimg & 7) * 8;
  const float* bh = bias16 + h * 4096;
#pragma unroll
  for (int ti = 0; ti < 4; ++ti) {
#pragma unroll
    for (int r = 0; r < 8; ++r) {
      int i = ti * 16 + hh * 8 + r;
      int ci = region9(whY + (i >> 3)) * 3 + region9(wwX + (i & 7));
      float sc[4];
#pragma unroll
      for (int tj = 0; tj < 4; ++tj) {
        int j = tj * 16 + cl;
        int cj = region9(whY + (j >> 3)) * 3 + region9(wwX + (j & 7));
        float v = acc[ti][tj][r] * scl + bh[i * 64 + j];
        if (ci != cj) v -= 100.0f;
        sc[tj] = v;
      }
      float m = fmaxf(fmaxf(sc[0], sc[1]), fmaxf(sc[2], sc[3]));
      for (int o = 1; o < 16; o <<= 1) m = fmaxf(m, __shfl_xor(m, o, 32));
      float sum = 0.f;
#pragma unroll
      for (int tj = 0; tj < 4; ++tj) { sc[tj] = __expf(sc[tj] - m); sum += sc[tj]; }
      for (int o = 1; o < 16; o <<= 1) sum += __shfl_xor(sum, o, 32);
      float inv = 1.0f / sum;
#pragma unroll
      for (int tj = 0; tj < 4; ++tj)
        probs[h * 4096 + i * 64 + tj * 16 + cl] = (__bf16)(sc[tj] * inv);
    }
  }
  __syncthreads();

  // ---- P @ V ----
  v8f o[4][2] = {};
  const __bf16* vb = vp + base * 64 * HD_;
#pragma unroll
  for (int s = 0; s < 2; ++s) {
    v16bf bv[2];
#pragma unroll
    for (int tv = 0; tv < 2; ++tv)
      bv[tv] = ld_bfrag(vb + (size_t)(s * 32) * HD_ + tv * 16 + cl, koff, HD_);
#pragma unroll
    for (int ti = 0; ti < 4; ++ti) {
      v16bf ap = ld_afrag(probs + h * 4096 + (ti * 16 + cl) * 64 + s * 32, koff);
#pragma unroll
      for (int tv = 0; tv < 2; ++tv)
        o[ti][tv] = wmma_bf16(ap, bv[tv], o[ti][tv]);
    }
  }
#pragma unroll
  for (int ti = 0; ti < 4; ++ti)
#pragma unroll
    for (int tv = 0; tv < 2; ++tv)
#pragma unroll
      for (int r = 0; r < 8; ++r) {
        int tok = ti * 16 + hh * 8 + r;
        int col = h * HD_ + tv * 16 + cl;
        attnout[((size_t)win * 64 + tok) * DIM_ + col] = (__bf16)o[ti][tv][r];
      }
}

// ---------------------------------------------------------------------------
// K7: x1 = shortcut + LN(y); also emit x1 in bf16 for the MLP GEMM.
// 1 wave per token (256-wide LN with full-wave shuffle reduce).
// ---------------------------------------------------------------------------
__global__ __launch_bounds__(256) void ln_add_kernel(
    const float* __restrict__ xin, const float* __restrict__ y,
    const float* __restrict__ w, const float* __restrict__ b,
    float* __restrict__ x1f, __bf16* __restrict__ x1b) {
  int wave = threadIdx.x >> 5, lane = threadIdx.x & 31;
  size_t row = (size_t)blockIdx.x * 8 + wave;
  const float* yr = y + row * DIM_;
  float v[8], s = 0.f;
#pragma unroll
  for (int j = 0; j < 8; ++j) { v[j] = yr[j * 32 + lane]; s += v[j]; }
  for (int o = 16; o; o >>= 1) s += __shfl_xor(s, o, 32);
  float mu = s * (1.0f / DIM_);
  float vs = 0.f;
#pragma unroll
  for (int j = 0; j < 8; ++j) { float d = v[j] - mu; vs += d * d; }
  for (int o = 16; o; o >>= 1) vs += __shfl_xor(vs, o, 32);
  float rs = rsqrtf(vs * (1.0f / DIM_) + 1e-5f);
#pragma unroll
  for (int j = 0; j < 8; ++j) {
    int c = j * 32 + lane;
    float ov = (v[j] - mu) * rs * w[c] + b[c] + xin[row * DIM_ + c];
    x1f[row * DIM_ + c] = ov;
    x1b[row * DIM_ + c] = (__bf16)ov;
  }
}

// K10: out = shortcut + x1 + LN(h2)
__global__ __launch_bounds__(256) void final_kernel(
    const float* __restrict__ xin, const float* __restrict__ x1,
    const float* __restrict__ h2, const float* __restrict__ w,
    const float* __restrict__ b, float* __restrict__ out) {
  int wave = threadIdx.x >> 5, lane = threadIdx.x & 31;
  size_t row = (size_t)blockIdx.x * 8 + wave;
  const float* hr = h2 + row * DIM_;
  float v[8], s = 0.f;
#pragma unroll
  for (int j = 0; j < 8; ++j) { v[j] = hr[j * 32 + lane]; s += v[j]; }
  for (int o = 16; o; o >>= 1) s += __shfl_xor(s, o, 32);
  float mu = s * (1.0f / DIM_);
  float vs = 0.f;
#pragma unroll
  for (int j = 0; j < 8; ++j) { float d = v[j] - mu; vs += d * d; }
  for (int o = 16; o; o >>= 1) vs += __shfl_xor(vs, o, 32);
  float rs = rsqrtf(vs * (1.0f / DIM_) + 1e-5f);
#pragma unroll
  for (int j = 0; j < 8; ++j) {
    int c = j * 32 + lane;
    out[row * DIM_ + c] =
        xin[row * DIM_ + c] + x1[row * DIM_ + c] + (v[j] - mu) * rs * w[c] + b[c];
  }
}

// ---------------------------------------------------------------------------
extern "C" void kernel_launch(void* const* d_in, const int* in_sizes, int n_in,
                              void* d_out, int out_size, void* d_ws, size_t ws_size,
                              hipStream_t stream) {
  const float* x        = (const float*)d_in[0];
  const float* qkv_w    = (const float*)d_in[1];
  const float* q_bias   = (const float*)d_in[2];
  const float* v_bias   = (const float*)d_in[3];
  const float* proj_w   = (const float*)d_in[4];
  const float* proj_b   = (const float*)d_in[5];
  const float* logit_sc = (const float*)d_in[6];
  const float* cpb_w1   = (const float*)d_in[7];
  const float* cpb_b1   = (const float*)d_in[8];
  const float* cpb_w2   = (const float*)d_in[9];
  const float* norm1_w  = (const float*)d_in[10];
  const float* norm1_b  = (const float*)d_in[11];
  const float* norm2_w  = (const float*)d_in[12];
  const float* norm2_b  = (const float*)d_in[13];
  const float* mlp_w1   = (const float*)d_in[14];
  const float* mlp_b1   = (const float*)d_in[15];
  const float* mlp_w2   = (const float*)d_in[16];
  const float* mlp_b2   = (const float*)d_in[17];
  float* out = (float*)d_out;

  char* ws = (char*)d_ws;
  size_t off = 0;
  auto take = [&](size_t bytes) -> void* {
    void* p = ws + off;
    off = (off + bytes + 255) & ~(size_t)255;
    return p;
  };

  float*  bias16 = (float*)  take((size_t)NH_ * 4096 * 4);
  float*  scaleb = (float*)  take(64);
  float*  qkvb   = (float*)  take(768 * 4);
  __bf16* Bqkv   = (__bf16*) take((size_t)196608 * 2);
  __bf16* Bproj  = (__bf16*) take((size_t)65536 * 2);
  __bf16* Bm1    = (__bf16*) take((size_t)262144 * 2);
  __bf16* Bm2    = (__bf16*) take((size_t)262144 * 2);
  __bf16* xw     = (__bf16*) take((size_t)TOK_ * DIM_ * 2);
  float*  qkvbuf = (float*)  take((size_t)TOK_ * 768 * 4);
  __bf16* qn     = (__bf16*) take((size_t)NWIN_ * NH_ * 64 * HD_ * 2);
  __bf16* knT    = (__bf16*) take((size_t)NWIN_ * NH_ * 64 * HD_ * 2);
  __bf16* vp     = (__bf16*) take((size_t)NWIN_ * NH_ * 64 * HD_ * 2);
  __bf16* attno  = (__bf16*) take((size_t)TOK_ * DIM_ * 2);
  float*  ybuf   = (float*)  take((size_t)TOK_ * DIM_ * 4);
  float*  x1f    = (float*)  take((size_t)TOK_ * DIM_ * 4);
  __bf16* x1b    = (__bf16*) take((size_t)TOK_ * DIM_ * 2);
  __bf16* h1     = (__bf16*) take((size_t)TOK_ * HID_ * 2);
  float*  h2     = (float*)  take((size_t)TOK_ * DIM_ * 4);
  (void)ws_size; (void)in_sizes; (void)n_in; (void)out_size;

  // allow 64 KB dynamic LDS for the fused attention kernel (idempotent)
  (void)hipFuncSetAttribute((const void*)attn_kernel,
                            hipFuncAttributeMaxDynamicSharedMemorySize,
                            NH_ * N_ * N_ * 2);

  cpb_kernel<<<1, 256, 0, stream>>>(cpb_w1, cpb_b1, cpb_w2, bias16);
  pack_kernel<<<3072, 256, 0, stream>>>(qkv_w, proj_w, mlp_w1, mlp_w2,
                                        q_bias, v_bias, logit_sc,
                                        Bqkv, Bproj, Bm1, Bm2, qkvb, scaleb);
  shiftwin_kernel<<<TOK_ * DIM_ / 256, 256, 0, stream>>>(x, xw);

  // QKV: [65536,256] @ [256,768] + bias -> f32
  gemm_kernel<0><<<(TOK_ / 64) * (768 / 128), 256, 0, stream>>>(
      xw, Bqkv, qkvb, qkvbuf, nullptr, TOK_, 768, DIM_, 768 / 128);

  normpack_kernel<<<NWIN_ * NH_ * 64 / 256, 256, 0, stream>>>(qkvbuf, qn, knT, vp);

  attn_kernel<<<NWIN_, 256, NH_ * N_ * N_ * 2, stream>>>(
      qn, knT, vp, bias16, scaleb, attno);

  // proj: [65536,256] @ [256,256] + b, fused window-reverse + roll scatter
  gemm_kernel<3><<<(TOK_ / 64) * (DIM_ / 128), 256, 0, stream>>>(
      attno, Bproj, proj_b, ybuf, nullptr, TOK_, DIM_, DIM_, DIM_ / 128);

  ln_add_kernel<<<TOK_ / 8, 256, 0, stream>>>(x, ybuf, norm1_w, norm1_b, x1f, x1b);

  // MLP1: [65536,256] @ [256,1024] + b -> exact GELU -> bf16
  gemm_kernel<2><<<(TOK_ / 64) * (HID_ / 128), 256, 0, stream>>>(
      x1b, Bm1, mlp_b1, nullptr, h1, TOK_, HID_, DIM_, HID_ / 128);

  // MLP2: [65536,1024] @ [1024,256] + b -> f32
  gemm_kernel<0><<<(TOK_ / 64) * (DIM_ / 128), 256, 0, stream>>>(
      h1, Bm2, mlp_b2, h2, nullptr, TOK_, DIM_, HID_, DIM_ / 128);

  final_kernel<<<TOK_ / 8, 256, 0, stream>>>(x, x1f, h2, norm2_w, norm2_b, out);
}
